// ResGINNet_60722247631482
// MI455X (gfx1250) — compile-verified, matrix operand order checked by reading, not run on hardware
//
#include <hip/hip_runtime.h>
#include <math.h>

typedef __attribute__((ext_vector_type(16))) _Float16 v16h;
typedef __attribute__((ext_vector_type(2)))  _Float16 h2;
typedef __attribute__((ext_vector_type(8)))  float    v8f;
typedef __attribute__((ext_vector_type(4)))  unsigned int u32x4;
typedef __attribute__((ext_vector_type(8)))  int      i32x8;
typedef __attribute__((ext_vector_type(4)))  int      i32x4;

#define NN   51200
#define NE   204800
#define Bsz  256
#define Tt   1000
#define DIMc 32
#define FXDc 78

#if __has_builtin(__builtin_amdgcn_tensor_load_to_lds) && __has_builtin(__builtin_amdgcn_s_wait_tensorcnt)
#define HAVE_TDM 1
#else
#define HAVE_TDM 0
#endif

// ---------------- WMMA fragment helpers (ISA 7.12.2 layouts) ----------------

// A (16x32 f16, row-major, arbitrary row stride). Also valid as B loader when
// the source is row-major [N,K] (torch-style W[out,in] == B^T).
__device__ inline v16h frag_rm_f16(const _Float16* __restrict__ p0, size_t row_stride, int lane) {
  const int r  = lane & 15;
  const int k0 = (lane & 16) ? 8 : 0;
  const _Float16* p = p0 + (size_t)r * row_stride + k0;
  v16h v;
#pragma unroll
  for (int j = 0; j < 8; ++j) { v[j] = p[j]; v[8 + j] = p[16 + j]; }
  return v;
}

// B (32x16) from row-major f32 [K,N]; converts to f16 in registers.
__device__ inline v16h fragB_kmaj_f32(const float* __restrict__ W, int N, int kk, int n0, int lane) {
  const int n  = n0 + (lane & 15);
  const int k0 = (lane & 16) ? 8 : 0;
  v16h v;
#pragma unroll
  for (int j = 0; j < 8; ++j) {
    v[j]     = (_Float16)W[(size_t)(kk + k0 + j)      * N + n];
    v[8 + j] = (_Float16)W[(size_t)(kk + 16 + k0 + j) * N + n];
  }
  return v;
}

__device__ inline v8f wmma32(v16h a, v16h b, v8f c) {
  return __builtin_amdgcn_wmma_f32_16x16x32_f16(false, a, false, b, (short)0, c, false, false);
}

__device__ inline float sigf(float x) { return 1.f / (1.f + __expf(-x)); }
__device__ inline unsigned rfl(unsigned x) { return (unsigned)__builtin_amdgcn_readfirstlane((int)x); }

#if HAVE_TDM
// TDM: async 2D tile load  global[256 rows x KX f16, row stride T*KX]  ->  LDS.
// Descriptor per CDNA5 ISA (group0 + group1, groups 2/3 zero).
__device__ inline void tdm_issue_x(const _Float16* X, const _Float16* ldst, int T, int KX, int t) {
  unsigned lds_addr = (unsigned)(size_t)ldst;                      // LDS byte offset
  unsigned long long ga = (unsigned long long)(size_t)(X + (size_t)t * KX);
  unsigned long long stride = (unsigned long long)T * (unsigned long long)KX; // elements

  u32x4 g0;
  g0[0] = rfl(1u);                                                  // count=1
  g0[1] = rfl(lds_addr);                                            // lds_addr
  g0[2] = rfl((unsigned)ga);                                        // global_addr[31:0]
  g0[3] = rfl(((unsigned)(ga >> 32) & 0x01FFFFFFu) | (2u << 30));   // ga[56:32] | type=2

  i32x8 g1;
  g1[0] = (int)rfl(0x00010000u);                                    // data_size=1 (2B)
  g1[1] = (int)rfl(((unsigned)KX & 0xFFFFu) << 16);                 // tensor_dim0 lo16
  g1[2] = (int)rfl((((unsigned)KX >> 16) & 0xFFFFu) | (256u << 16));// dim0 hi | tensor_dim1 lo
  g1[3] = (int)rfl(((unsigned)KX & 0xFFFFu) << 16);                 // dim1 hi(0) | tile_dim0
  g1[4] = (int)rfl(256u);                                           // tile_dim1=256, tile_dim2=0
  g1[5] = (int)rfl((unsigned)stride);                               // dim0_stride lo32
  g1[6] = (int)rfl((unsigned)((stride >> 32) & 0xFFFFu));           // dim0_stride hi16
  g1[7] = 0;

  i32x4 gz; gz[0] = gz[1] = gz[2] = gz[3] = 0;
#if __clang_major__ >= 23
  i32x8 gz8;
#pragma unroll
  for (int j = 0; j < 8; ++j) gz8[j] = 0;
  __builtin_amdgcn_tensor_load_to_lds(g0, g1, gz, gz, gz8, 0);
#else
  __builtin_amdgcn_tensor_load_to_lds(g0, g1, gz, gz, 0);
#endif
}
#endif

// ---------------- small utility kernels ----------------

__global__ void zero_f32(float* p, long n) {
  long i = (long)blockIdx.x * blockDim.x + threadIdx.x;
  if (i < n) p[i] = 0.f;
}

__global__ void cvt_f16(const float* __restrict__ s, _Float16* __restrict__ d, long n) {
  long i = (long)blockIdx.x * blockDim.x + threadIdx.x;
  if (i < n) d[i] = (_Float16)s[i];
}

__global__ void emb_kernel(const float* __restrict__ emb, const int* __restrict__ target,
                           _Float16* __restrict__ xemb, long total) {
  long i = (long)blockIdx.x * blockDim.x + threadIdx.x;
  if (i >= total) return;
  long bt = i >> 7; int e = (int)(i & 127);
  xemb[i] = (_Float16)emb[(size_t)target[bt] * 128 + e];
}

// ---------------- GIN kernels ----------------

__global__ void gather_kernel(const float* __restrict__ h, const int* __restrict__ src,
                              const int* __restrict__ dst, float* __restrict__ agg,
                              int kin, long total) {
  long i = (long)blockIdx.x * blockDim.x + threadIdx.x;
  if (i >= total) return;
  long e = i / kin; int c = (int)(i - e * kin);
  atomicAdd(&agg[(size_t)dst[e] * kin + c], h[(size_t)src[e] * kin + c]);
}

__global__ void mlp_kernel(const float* __restrict__ hin, const float* __restrict__ agg,
                           const float* __restrict__ W, const float* __restrict__ b,
                           float* __restrict__ out, int kin, int kout,
                           int add_agg, int do_relu, long total) {
  long i = (long)blockIdx.x * blockDim.x + threadIdx.x;
  if (i >= total) return;
  long n = i / kout; int j = (int)(i - n * kout);
  float s = b[j];
  const float* hr = hin + (size_t)n * kin;
  const float* ar = agg + (size_t)n * kin;
  for (int k = 0; k < kin; ++k) {
    float z = add_agg ? (hr[k] + ar[k]) : hr[k];
    s += z * W[(size_t)k * kout + j];
  }
  out[i] = do_relu ? fmaxf(s, 0.f) : s;
}

__global__ void bn_stats_kernel(const float* __restrict__ r, float* __restrict__ mean,
                                float* __restrict__ var) {
  __shared__ float s1[256], s2[256];
  const int ch = blockIdx.x;
  float a = 0.f, b = 0.f;
  for (int n = threadIdx.x; n < NN; n += 256) {
    float v = r[(size_t)n * DIMc + ch];
    a += v; b += v * v;
  }
  s1[threadIdx.x] = a; s2[threadIdx.x] = b;
  __syncthreads();
  for (int w = 128; w > 0; w >>= 1) {
    if (threadIdx.x < w) { s1[threadIdx.x] += s1[threadIdx.x + w]; s2[threadIdx.x] += s2[threadIdx.x + w]; }
    __syncthreads();
  }
  if (threadIdx.x == 0) {
    float m = s1[0] / (float)NN;
    mean[ch] = m;
    var[ch]  = s2[0] / (float)NN - m * m;
  }
}

__global__ void bn_norm_kernel(const float* __restrict__ r, const float* __restrict__ mean,
                               const float* __restrict__ var, const float* __restrict__ g,
                               const float* __restrict__ be, float* __restrict__ hout,
                               float* __restrict__ hs, int layer, long total) {
  long i = (long)blockIdx.x * blockDim.x + threadIdx.x;
  if (i >= total) return;
  long n = i >> 5; int ch = (int)(i & 31);
  float v = g[ch] * (r[i] - mean[ch]) * rsqrtf(var[ch] + 1e-5f) + be[ch];
  hout[i] = v;
  hs[(size_t)n * 160 + layer * 32 + ch] = v;
}

__global__ void pool_kernel(const float* __restrict__ hs, const int* __restrict__ batch,
                            float* __restrict__ pooled, long total) {
  long i = (long)blockIdx.x * blockDim.x + threadIdx.x;
  if (i >= total) return;
  long n = i / 160; int j = (int)(i - n * 160);
  atomicAdd(&pooled[(size_t)batch[n] * 160 + j], hs[i]);
}

__global__ void dense_kernel(const float* __restrict__ A, const float* __restrict__ W,
                             const float* __restrict__ b, float* __restrict__ out,
                             int M, int K, int N, int ldo, int do_relu) {
  long i = (long)blockIdx.x * blockDim.x + threadIdx.x;
  if (i >= (long)M * N) return;
  int row = (int)(i / N), col = (int)(i - (long)row * N);
  float s = b[col];
  const float* ar = A + (size_t)row * K;
  for (int k = 0; k < K; ++k) s += ar[k] * W[(size_t)k * N + col];
  out[(size_t)row * ldo + col] = do_relu ? fmaxf(s, 0.f) : s;
}

// ---------------- persistent bidirectional LSTM layer ----------------
// One 512-thread block per direction. h state: LDS f16 (64 KB). c state + bias:
// VGPR-resident (wave owns fixed 16x16 patches so i/f/g/o for a cell co-reside).
// X_t tiles double-buffered in LDS, filled by the Tensor Data Mover one step
// ahead (TENSORcnt + barrier ordering). 16 waves x 8 patches = 128 gate patches.
__global__ void __launch_bounds__(512) lstm_dir_kernel(
    const _Float16* __restrict__ X, const _Float16* __restrict__ WihAll,
    const _Float16* __restrict__ WhhAll, const float* __restrict__ biasAll,
    _Float16* __restrict__ Y, int T, int KX) {
  const int d = blockIdx.x;
  const _Float16* Wih  = WihAll + (size_t)d * 512 * KX;
  const _Float16* Whh  = WhhAll + (size_t)d * 512 * 128;
  const float*    bias = biasAll + d * 512;

  extern __shared__ _Float16 smem[];
  _Float16* hbuf = smem;                    // [256][128]
  _Float16* xs   = smem + 256 * 128;        // [2][256][KX]

  const int lane = threadIdx.x & 31;
  const int wave = threadIdx.x >> 5;        // 0..15

  for (int i = threadIdx.x; i < 256 * 128; i += 512) hbuf[i] = (_Float16)0.f;

  float creg[8][8];
  float breg[8][4];
#pragma unroll
  for (int p = 0; p < 8; ++p) {
    const int patch = wave * 8 + p;
    const int c0 = (patch & 7) * 16;
#pragma unroll
    for (int r = 0; r < 8; ++r) creg[p][r] = 0.f;
#pragma unroll
    for (int g = 0; g < 4; ++g) breg[p][g] = bias[g * 128 + c0 + (lane & 15)];
  }

  int cur = 0;
#if HAVE_TDM
  if (threadIdx.x == 0) {
    tdm_issue_x(X, xs, T, KX, d ? T - 1 : 0);
    __builtin_amdgcn_s_wait_tensorcnt(0);
  }
#endif
  __syncthreads();

  for (int step = 0; step < T; ++step) {
    const int t = d ? (T - 1 - step) : step;
#if HAVE_TDM
    if (threadIdx.x == 0 && step + 1 < T)   // prefetch X(t+1) into the idle buffer
      tdm_issue_x(X, xs + (size_t)(cur ^ 1) * 256 * KX, T, KX, d ? t - 1 : t + 1);
#endif
    h2 hv[8][4];
#pragma unroll
    for (int p = 0; p < 8; ++p) {
      const int patch = wave * 8 + p;
      const int m0 = (patch >> 3) * 16;
      const int c0 = (patch & 7) * 16;
      v8f acc[4];
#pragma unroll
      for (int g = 0; g < 4; ++g)
#pragma unroll
        for (int j = 0; j < 8; ++j) acc[g][j] = breg[p][g];
      // input contribution: X_t @ Wih^T
      for (int kk = 0; kk < KX; kk += 32) {
#if HAVE_TDM
        v16h a = frag_rm_f16(xs + (size_t)cur * 256 * KX + (size_t)m0 * KX + kk, (size_t)KX, lane);
#else
        v16h a = frag_rm_f16(X + ((size_t)m0 * T + t) * KX + kk, (size_t)T * KX, lane);
#endif
#pragma unroll
        for (int g = 0; g < 4; ++g) {
          v16h b = frag_rm_f16(Wih + (size_t)(g * 128 + c0) * KX + kk, (size_t)KX, lane);
          acc[g] = wmma32(a, b, acc[g]);
        }
      }
      // recurrent contribution: h @ Whh^T (h from LDS)
#pragma unroll
      for (int kk = 0; kk < 128; kk += 32) {
        v16h a = frag_rm_f16(hbuf + (size_t)m0 * 128 + kk, 128, lane);
#pragma unroll
        for (int g = 0; g < 4; ++g) {
          v16h b = frag_rm_f16(Whh + (size_t)(g * 128 + c0) * 128 + kk, 128, lane);
          acc[g] = wmma32(a, b, acc[g]);
        }
      }
      // torch gate order i,f,g,o
#pragma unroll
      for (int r = 0; r < 8; ++r) {
        float ig = sigf(acc[0][r]);
        float fg = sigf(acc[1][r]);
        float gg = tanhf(acc[2][r]);
        float og = sigf(acc[3][r]);
        float cc = fg * creg[p][r] + ig * gg;
        creg[p][r] = cc;
        hv[p][r >> 1][r & 1] = (_Float16)(og * tanhf(cc));
      }
    }
    __syncthreads();                        // all LDS reads of h(t) complete
#pragma unroll
    for (int p = 0; p < 8; ++p) {
      const int patch = wave * 8 + p;
      const int m0 = (patch >> 3) * 16;
      const int c0 = (patch & 7) * 16;
      const int n  = c0 + (lane & 15);
#pragma unroll
      for (int r = 0; r < 8; ++r) {
        const int m = m0 + r + ((lane & 16) ? 8 : 0);
        _Float16 hh = hv[p][r >> 1][r & 1];
        hbuf[(size_t)m * 128 + n] = hh;
        Y[((size_t)m * T + t) * 256 + d * 128 + n] = hh;
      }
    }
#if HAVE_TDM
    if (threadIdx.x == 0) __builtin_amdgcn_s_wait_tensorcnt(0); // X(t+1) landed
#endif
    __syncthreads();
    cur ^= 1;
  }
}

// ---------------- fcxt: [256,256000](f16) @ [256000,128](f32->f16) ----------------
__global__ void __launch_bounds__(256) fcxt_kernel(const _Float16* __restrict__ Xf,
                                                   const float* __restrict__ W,
                                                   const float* __restrict__ bias,
                                                   float* __restrict__ xc, int Ktot) {
  const int lane = threadIdx.x & 31;
  const int wave = threadIdx.x >> 5;        // 8 waves -> 8 N-tiles (N=128)
  const int m0 = blockIdx.x * 16;
  const int n0 = wave * 16;
  v8f acc;
#pragma unroll
  for (int j = 0; j < 8; ++j) acc[j] = 0.f;
  for (int kk = 0; kk < Ktot; kk += 32) {
    if (kk + 96 < Ktot) __builtin_prefetch(&W[(size_t)(kk + 96) * 128], 0, 0);
    v16h a = frag_rm_f16(Xf + (size_t)m0 * Ktot + kk, (size_t)Ktot, lane);
    v16h b = fragB_kmaj_f32(W, 128, kk, n0, lane);
    acc = wmma32(a, b, acc);
  }
  const int n = n0 + (lane & 15);
  const float bv = bias[n];
#pragma unroll
  for (int r = 0; r < 8; ++r) {
    const int m = m0 + r + ((lane & 16) ? 8 : 0);
    xc[(size_t)m * 256 + 128 + n] = acc[r] + bv;
  }
}

// ---------------- host orchestration ----------------

static inline int gb(long n, int t) { return (int)((n + t - 1) / t); }

extern "C" void kernel_launch(void* const* d_in, const int* in_sizes, int n_in,
                              void* d_out, int out_size, void* d_ws, size_t ws_size,
                              hipStream_t stream) {
  const float* x      = (const float*)d_in[0];
  const int*   edge   = (const int*)d_in[1];
  const int*   batch  = (const int*)d_in[2];
  const int*   target = (const int*)d_in[3];
  const float* g1_W1  = (const float*)d_in[6];
  const float* g1_b1  = (const float*)d_in[7];
  const float* g1_W2  = (const float*)d_in[8];
  const float* g1_b2  = (const float*)d_in[9];
  const float* gin_Wa = (const float*)d_in[10];
  const float* gin_ba = (const float*)d_in[11];
  const float* gin_Wb = (const float*)d_in[12];
  const float* gin_bb = (const float*)d_in[13];
  const float* bn_g   = (const float*)d_in[14];
  const float* bn_b   = (const float*)d_in[15];
  const float* fcg1_W = (const float*)d_in[16];
  const float* fcg1_b = (const float*)d_in[17];
  const float* emb    = (const float*)d_in[18];
  const float* Wih0   = (const float*)d_in[19];
  const float* Whh0   = (const float*)d_in[20];
  const float* b0     = (const float*)d_in[21];
  const float* Wih1   = (const float*)d_in[22];
  const float* Whh1   = (const float*)d_in[23];
  const float* b1     = (const float*)d_in[24];
  const float* fcxt_W = (const float*)d_in[25];
  const float* fcxt_b = (const float*)d_in[26];
  const float* fc1_W  = (const float*)d_in[27];
  const float* fc1_b  = (const float*)d_in[28];
  const float* fc2_W  = (const float*)d_in[29];
  const float* fc2_b  = (const float*)d_in[30];
  const float* out_W  = (const float*)d_in[31];
  const float* out_b  = (const float*)d_in[32];

  const int* src = edge;
  const int* dst = edge + NE;

  size_t off = 0;
  auto take = [&](size_t bytes) -> void* {
    void* p = (char*)d_ws + off;
    off += (bytes + 255) & ~(size_t)255;
    return p;
  };
  float* agg    = (float*)take((size_t)NN * 78 * 4);
  float* hcur   = (float*)take((size_t)NN * 32 * 4);
  float* tbuf   = (float*)take((size_t)NN * 32 * 4);
  float* rbuf   = (float*)take((size_t)NN * 32 * 4);
  float* hsb    = (float*)take((size_t)NN * 160 * 4);
  float* bnm    = (float*)take(32 * 4);
  float* bnv    = (float*)take(32 * 4);
  float* pooled = (float*)take((size_t)Bsz * 160 * 4);
  float* xc     = (float*)take((size_t)Bsz * 256 * 4);
  float* a1     = (float*)take((size_t)Bsz * 1024 * 4);
  float* a2     = (float*)take((size_t)Bsz * 512 * 4);
  _Float16* xemb  = (_Float16*)take((size_t)Bsz * Tt * 128 * 2);
  _Float16* wih0h = (_Float16*)take((size_t)2 * 512 * 128 * 2);
  _Float16* whh0h = (_Float16*)take((size_t)2 * 512 * 128 * 2);
  _Float16* wih1h = (_Float16*)take((size_t)2 * 512 * 256 * 2);
  _Float16* whh1h = (_Float16*)take((size_t)2 * 512 * 128 * 2);
  _Float16* out0  = (_Float16*)take((size_t)Bsz * Tt * 256 * 2);
  _Float16* out1  = (_Float16*)take((size_t)Bsz * Tt * 256 * 2);

  // ---- GIN stack (5 layers) ----
  for (int l = 0; l < 5; ++l) {
    const int kin = (l == 0) ? FXDc : DIMc;
    const float* hin = (l == 0) ? x : hcur;
    const float* Wa  = (l == 0) ? g1_W1 : gin_Wa + (size_t)(l - 1) * 32 * 32;
    const float* ba  = (l == 0) ? g1_b1 : gin_ba + (size_t)(l - 1) * 32;
    const float* Wb  = (l == 0) ? g1_W2 : gin_Wb + (size_t)(l - 1) * 32 * 32;
    const float* bb  = (l == 0) ? g1_b2 : gin_bb + (size_t)(l - 1) * 32;

    long na = (long)NN * kin;
    zero_f32<<<gb(na, 256), 256, 0, stream>>>(agg, na);
    long ne = (long)NE * kin;
    gather_kernel<<<gb(ne, 256), 256, 0, stream>>>(hin, src, dst, agg, kin, ne);
    long n1 = (long)NN * 32;
    mlp_kernel<<<gb(n1, 256), 256, 0, stream>>>(hin, agg, Wa, ba, tbuf, kin, 32, 1, 1, n1);
    mlp_kernel<<<gb(n1, 256), 256, 0, stream>>>(tbuf, agg, Wb, bb, rbuf, 32, 32, 0, 1, n1);
    bn_stats_kernel<<<32, 256, 0, stream>>>(rbuf, bnm, bnv);
    bn_norm_kernel<<<gb(n1, 256), 256, 0, stream>>>(rbuf, bnm, bnv, bn_g + l * 32, bn_b + l * 32,
                                                    hcur, hsb, l, n1);
  }

  // ---- graph pooling + fcg1 -> xc[:, 0:128] ----
  long np = (long)Bsz * 160;
  zero_f32<<<gb(np, 256), 256, 0, stream>>>(pooled, np);
  long nh = (long)NN * 160;
  pool_kernel<<<gb(nh, 256), 256, 0, stream>>>(hsb, batch, pooled, nh);
  dense_kernel<<<gb((long)Bsz * 128, 256), 256, 0, stream>>>(pooled, fcg1_W, fcg1_b, xc,
                                                             Bsz, 160, 128, 256, 1);

  // ---- embedding + weight conversion to f16 ----
  long nemb = (long)Bsz * Tt * 128;
  emb_kernel<<<gb(nemb, 256), 256, 0, stream>>>(emb, target, xemb, nemb);
  cvt_f16<<<gb(2L * 512 * 128, 256), 256, 0, stream>>>(Wih0, wih0h, 2L * 512 * 128);
  cvt_f16<<<gb(2L * 512 * 128, 256), 256, 0, stream>>>(Whh0, whh0h, 2L * 512 * 128);
  cvt_f16<<<gb(2L * 512 * 256, 256), 256, 0, stream>>>(Wih1, wih1h, 2L * 512 * 256);
  cvt_f16<<<gb(2L * 512 * 128, 256), 256, 0, stream>>>(Whh1, whh1h, 2L * 512 * 128);

  // ---- bidirectional LSTM layers (persistent WMMA+TDM kernels, 1 block/dir) ----
  size_t sh0 = (size_t)256 * 128 * 2 + (size_t)2 * 256 * 128 * 2;  // 192 KB
  size_t sh1 = (size_t)256 * 128 * 2 + (size_t)2 * 256 * 256 * 2;  // 320 KB
  lstm_dir_kernel<<<2, 512, sh0, stream>>>(xemb, wih0h, whh0h, b0, out0, Tt, 128);
  lstm_dir_kernel<<<2, 512, sh1, stream>>>(out0, wih1h, whh1h, b1, out1, Tt, 256);

  // ---- fcxt (memory-bound streaming WMMA) -> xc[:, 128:256] ----
  fcxt_kernel<<<16, 256, 0, stream>>>(out1, fcxt_W, fcxt_b, xc, Tt * 256);

  // ---- head MLP ----
  dense_kernel<<<gb((long)Bsz * 1024, 256), 256, 0, stream>>>(xc, fc1_W, fc1_b, a1,
                                                              Bsz, 256, 1024, 1024, 1);
  dense_kernel<<<gb((long)Bsz * 512, 256), 256, 0, stream>>>(a1, fc2_W, fc2_b, a2,
                                                             Bsz, 1024, 512, 512, 1);
  dense_kernel<<<gb((long)Bsz, 256), 256, 0, stream>>>(a2, out_W, out_b, (float*)d_out,
                                                       Bsz, 512, 1, 1, 0);
}
